// FunctaTransformer_3788161155737
// MI455X (gfx1250) — matrix operand order repaired and checked by reading, not compiled, not run on hardware
//
#include <hip/hip_runtime.h>
#include <hip/hip_bf16.h>

// ---------------------------------------------------------------------------
// Swin-style block for MI455X (gfx1250, wave32, WMMA f16 -> f32 accum,
// Tensor Data Mover staging for f16 operand tiles).
// ---------------------------------------------------------------------------

typedef __attribute__((ext_vector_type(2)))  _Float16 h2;
typedef __attribute__((ext_vector_type(8)))  _Float16 h8;
typedef __attribute__((ext_vector_type(16))) _Float16 h16;
typedef __attribute__((ext_vector_type(8)))  float    f8;
typedef __attribute__((ext_vector_type(4)))  unsigned u32x4;
typedef __attribute__((ext_vector_type(8)))  int      i32x8;
typedef __attribute__((ext_vector_type(4)))  int      i32x4;

#define Bdim  16
#define Edim  384
#define Ddim  64
#define Hnum  12
#define DH    32
#define Sdim  64      // tokens per window
#define NWdim 64      // windows per image
#define HID   1536
#define NTOK  65536   // B * D1 * D2

#if defined(__has_builtin)
#  if __has_builtin(__builtin_amdgcn_tensor_load_to_lds)
#    define HAVE_TDM 1
#  endif
#endif
#ifndef HAVE_TDM
#  define HAVE_TDM 0
#endif

__device__ __forceinline__ h16 cat16(h8 lo, h8 hi) {
  return __builtin_shufflevector(lo, hi, 0,1,2,3,4,5,6,7,8,9,10,11,12,13,14,15);
}

__device__ __forceinline__ float gelu_exact(float x) {
  return 0.5f * x * (1.0f + erff(x * 0.70710678118654752f));
}

#if HAVE_TDM
// ---------------------------------------------------------------------------
// TDM: stage a 2-D f16 tile (rows x 32 halves, 64B per row) into LDS with a
// 16B pad after every row (pad_interval = 16 DWORDs, pad_amount = 4 DWORDs),
// i.e. LDS row stride = 40 halves = 80B.  D# per ISA 8.3/8.4; groups 2/3
// zeroed (tensor <= 2D).  Issue from ONE wave; pair with s_wait_tensorcnt.
// ---------------------------------------------------------------------------
__device__ __forceinline__ void tdm_load_tile64x32(const _Float16* gsrc,
                                                   unsigned ldsByteAddr,
                                                   int rowStrideElems)
{
  const unsigned long long ga = (unsigned long long)(uintptr_t)gsrc;
  u32x4 g0;
  g0[0] = 1u;                                               // count=1, user D#
  g0[1] = ldsByteAddr;                                      // lds_addr
  g0[2] = (unsigned)(ga & 0xffffffffull);                   // global_addr lo
  g0[3] = (unsigned)((ga >> 32) & 0x1ffffffull) | (2u << 30); // hi + type=2
  i32x8 g1;
  g1[0] = (int)((1u << 16)      // data_size = 2 bytes
              | (1u << 20)      // pad_enable
              | (3u << 22)      // pad_interval: 16 DWORDs (64B) before pad
              | (3u << 25));    // pad_amount: 4 DWORDs (16B)
  g1[1] = (int)(((unsigned)rowStrideElems & 0xffffu) << 16); // tensor_dim0 lo16
  g1[2] = (int)(((unsigned)rowStrideElems >> 16) | (64u << 16)); // dim0 hi / dim1=64
  g1[3] = (int)(32u << 16);                                  // tile_dim0 = 32
  g1[4] = 64;                                                // tile_dim1 = 64
  g1[5] = rowStrideElems;                                    // dim0_stride lo32
  g1[6] = 0;
  g1[7] = 0;
  i32x4 gz = {0, 0, 0, 0};
#if __clang_major__ >= 23
  i32x8 gz8 = {0, 0, 0, 0, 0, 0, 0, 0};
  __builtin_amdgcn_tensor_load_to_lds(g0, g1, gz, gz, gz8, 0);
#else
  __builtin_amdgcn_tensor_load_to_lds(g0, g1, gz, gz, 0);
#endif
}
#endif

// ---------------------------------------------------------------------------
// Shared WMMA GEMM core: block computes a 64(M) x 64(N) tile of  A(f16) @ W(f32).
// 8 waves = 4(M) x 2(N-half); each wave owns two 16x16 WMMA tiles sharing one
// A fragment.  K multiple of 32.  A tile staged by the TDM (wave 0) into a
// 40-half-stride LDS buffer; W panel converted f32->f16 and transposed to
// [N][K] by VALU so B fragments are two 8-half reads along k.
// ---------------------------------------------------------------------------
struct AccPair { f8 a0, a1; };

__device__ AccPair gemm_core(const _Float16* __restrict__ A,
                             const float* __restrict__ W,
                             int K, int N, int m0, int n0,
                             _Float16* ldsA /*64*40*/, _Float16* ldsBT /*64*40*/)
{
  const int tid  = threadIdx.x;
  const int lane = tid & 31;
  const int wid  = tid >> 5;
  const int wm   = wid & 3;        // wave M tile (0..3)
  const int wn2  = wid >> 2;       // wave N half (0..1)

  // W loader: thread owns column n and k-pairs {kb, kb+1} + it*8
  const int wn_n = tid & 63;
  const int wk_b = (tid >> 6) * 2;

  // fragment coordinates (ISA 7.12.2 layouts)
  const int far  = wm * 16 + (lane & 15);
  const int fakh = (lane >> 4) << 3;   // 0 or 8
  const int fbc  = wn2 * 32 + (lane & 15);
  const int fbk  = (lane >> 4) << 3;   // 0 or 8

#if HAVE_TDM
  const unsigned ldsA_addr = (unsigned)(uintptr_t)(void*)ldsA;
#else
  const int arow = tid >> 2;
  const int acol = (tid & 3) << 3;
#endif

  f8 acc0 = {0.f,0.f,0.f,0.f,0.f,0.f,0.f,0.f};
  f8 acc1 = {0.f,0.f,0.f,0.f,0.f,0.f,0.f,0.f};

  for (int k0 = 0; k0 < K; k0 += 32) {
    // W panel into registers before the barrier (lanes contiguous in n)
    float wv0[4], wv1[4];
#pragma unroll
    for (int it = 0; it < 4; ++it) {
      int k = wk_b + it * 8;
      wv0[it] = W[(size_t)(k0 + k) * N + n0 + wn_n];
      wv1[it] = W[(size_t)(k0 + k + 1) * N + n0 + wn_n];
    }
    if (k0 + 32 < K)  // hint next W panel toward GL2 (global_prefetch_b8)
      __builtin_prefetch(&W[(size_t)(k0 + 32 + wk_b) * N + n0 + wn_n], 0, 1);
#if !HAVE_TDM
    h8 av8 = *(const h8*)(A + (size_t)(m0 + arow) * K + k0 + acol);
#endif

    __syncthreads();   // previous iteration's consumers are done

#if HAVE_TDM
    if (wid == 0)      // one wave drives the Tensor Data Mover for the A tile
      tdm_load_tile64x32(A + (size_t)m0 * K + k0, ldsA_addr, K);
#else
    *(h8*)(ldsA + arow * 40 + acol) = av8;
#endif
#pragma unroll
    for (int it = 0; it < 4; ++it) {
      int k = wk_b + it * 8;
      h2 p; p.x = (_Float16)wv0[it]; p.y = (_Float16)wv1[it];
      *(h2*)(ldsBT + wn_n * 40 + k) = p;
    }
#if HAVE_TDM
    if (wid == 0) __builtin_amdgcn_s_wait_tensorcnt(0);
#endif
    __syncthreads();

    h8 a0 = *(const h8*)(ldsA + far * 40 + fakh);
    h8 a1 = *(const h8*)(ldsA + far * 40 + fakh + 16);
    h16 av = cat16(a0, a1);
    h8 b00 = *(const h8*)(ldsBT + fbc * 40 + fbk);
    h8 b01 = *(const h8*)(ldsBT + fbc * 40 + fbk + 16);
    acc0 = __builtin_amdgcn_wmma_f32_16x16x32_f16(false, av, false, cat16(b00, b01),
                                                  (short)0, acc0, false, false);
    h8 b10 = *(const h8*)(ldsBT + (fbc + 16) * 40 + fbk);
    h8 b11 = *(const h8*)(ldsBT + (fbc + 16) * 40 + fbk + 16);
    acc1 = __builtin_amdgcn_wmma_f32_16x16x32_f16(false, av, false, cat16(b10, b11),
                                                  (short)0, acc1, false, false);
  }
  AccPair r; r.a0 = acc0; r.a1 = acc1;
  return r;
}

// ---------------------------------------------------------------------------
// K1: LayerNorm1 + cyclic shift(-4,-4) + window partition.
// ---------------------------------------------------------------------------
__global__ __launch_bounds__(256)
void k_ln1_window(const float* __restrict__ x, const float* __restrict__ g1,
                  const float* __restrict__ b1, float* __restrict__ xt,
                  _Float16* __restrict__ xw)
{
  const int blk = blockIdx.x;
  const int b   = blk >> 6;
  const int d1  = blk & 63;
  const int tid = threadIdx.x;
  const int d2  = tid & 63;
  const int eg  = tid >> 6;   // 0..3

  __shared__ float psum[4][64];
  __shared__ float psq [4][64];
  __shared__ float smu[64], srs[64];
  __shared__ float tile[64][33];

  const size_t xbase = ((size_t)b * Edim) * 4096 + (size_t)d1 * 64;

  float s = 0.f, sq = 0.f;
  for (int e = eg; e < Edim; e += 4) {
    float v = x[xbase + (size_t)e * 4096 + d2];
    s += v; sq += v * v;
  }
  psum[eg][d2] = s; psq[eg][d2] = sq;
  __syncthreads();
  if (tid < 64) {
    float ts = psum[0][tid] + psum[1][tid] + psum[2][tid] + psum[3][tid];
    float tq = psq [0][tid] + psq [1][tid] + psq [2][tid] + psq [3][tid];
    float mu  = ts * (1.0f / Edim);
    float var = tq * (1.0f / Edim) - mu * mu;
    smu[tid] = mu;
    srs[tid] = rsqrtf(var + 1e-5f);
  }
  __syncthreads();

  const int i = (d1 + 60) & 63;

  for (int e0 = 0; e0 < Edim; e0 += 32) {
    float vals[8];
#pragma unroll
    for (int ee = 0; ee < 8; ++ee)
      vals[ee] = x[xbase + (size_t)(e0 + eg * 8 + ee) * 4096 + d2];
    __syncthreads();
#pragma unroll
    for (int ee = 0; ee < 8; ++ee) tile[d2][eg * 8 + ee] = vals[ee];
    __syncthreads();

    const int tok = tid >> 2;
    const int off = (tid & 3) * 8;
    const float mu = smu[tok], rs = srs[tok];
    const size_t trow = ((size_t)b * 4096 + (size_t)d1 * 64 + tok) * Edim + e0 + off;
    const int j = (tok + 60) & 63;
    const int w = (i >> 3) * 8 + (j >> 3);
    const int sIdx = (i & 7) * 8 + (j & 7);
    const size_t wrow = (((size_t)b * NWdim + w) * Sdim + sIdx) * Edim + e0 + off;
#pragma unroll
    for (int q = 0; q < 8; ++q) {
      float v = tile[tok][off + q];
      xt[trow + q] = v;
      int e = e0 + off + q;
      xw[wrow + q] = (_Float16)((v - mu) * rs * g1[e] + b1[e]);
    }
  }
}

// ---------------------------------------------------------------------------
// K2: fused QKV projection.  grid = (M/64, E/64, 3), per-head output layout.
// ---------------------------------------------------------------------------
__global__ __launch_bounds__(256)
void k_qkv(const _Float16* __restrict__ xw,
           const float* __restrict__ Wq, const float* __restrict__ bq,
           const float* __restrict__ Wk, const float* __restrict__ bk,
           const float* __restrict__ Wv, const float* __restrict__ bv,
           _Float16* __restrict__ qo, _Float16* __restrict__ ko,
           _Float16* __restrict__ vo)
{
  __shared__ __align__(16) _Float16 ldsA[64 * 40];
  __shared__ __align__(16) _Float16 ldsBT[64 * 40];

  const int which = blockIdx.z;
  const float* W    = (which == 0) ? Wq : (which == 1) ? Wk : Wv;
  const float* bias = (which == 0) ? bq : (which == 1) ? bk : bv;
  _Float16*    out  = (which == 0) ? qo : (which == 1) ? ko : vo;
  const float scale = (which == 0) ? 0.17677669529663687f : 1.0f;

  const int m0 = blockIdx.x * 64;
  const int n0 = blockIdx.y * 64;

  AccPair r = gemm_core(xw, W, Edim, Edim, m0, n0, ldsA, ldsBT);

  const int lane = threadIdx.x & 31;
  const int wid  = threadIdx.x >> 5;
  const int wm = wid & 3, wn2 = wid >> 2;
  const int mbase = m0 + wm * 16 + ((lane >> 4) * 8);
#pragma unroll
  for (int nt = 0; nt < 2; ++nt) {
    f8 acc = nt ? r.a1 : r.a0;
    const int gn = n0 + wn2 * 32 + nt * 16 + (lane & 15);
    const float bb = bias[gn];
    const int h = gn >> 5, dh = gn & 31;
#pragma unroll
    for (int i = 0; i < 8; ++i) {
      int gm = mbase + i;
      int bw = gm >> 6, sIdx = gm & 63;
      float val = (acc[i] + bb) * scale;
      out[(((size_t)bw * Hnum + h) * Sdim + sIdx) * DH + dh] = (_Float16)val;
    }
  }
}

// ---------------------------------------------------------------------------
// K3: windowed attention.  One block per (b*w, head-pair); 4 waves per head.
// q/k tiles staged by the TDM (first wave of each half), v transposed by VALU.
// ---------------------------------------------------------------------------
__global__ __launch_bounds__(256)
void k_attn(const _Float16* __restrict__ qg, const _Float16* __restrict__ kg,
            const _Float16* __restrict__ vg, const float* __restrict__ pb,
            const float* __restrict__ mask, _Float16* __restrict__ z)
{
  const int idx = blockIdx.x;
  const int bw  = idx / 6;
  const int hp  = idx % 6;
  const int w   = bw & 63;
  const int tid = threadIdx.x;
  const int side = tid >> 7;           // which head of the pair
  const int h    = hp * 2 + side;
  const int lt   = tid & 127;
  const int lane = tid & 31;
  const int wv   = (tid >> 5) & 3;     // wave within half (0..3)

  __shared__ __align__(16) _Float16 sQ [2][64 * 40];
  __shared__ __align__(16) _Float16 sK [2][64 * 40];
  __shared__ __align__(16) _Float16 sVT[2][32 * 72];
  __shared__ __align__(16) float    sSC[2][64 * 68];

  const size_t hbase = (((size_t)bw * Hnum) + h) * (Sdim * DH);

  // stage q,k (TDM, padded rows) and v transposed (VALU)
  {
#if HAVE_TDM
    if (wv == 0) {
      tdm_load_tile64x32(qg + hbase, (unsigned)(uintptr_t)(void*)&sQ[side][0], 32);
      tdm_load_tile64x32(kg + hbase, (unsigned)(uintptr_t)(void*)&sK[side][0], 32);
    }
#endif
    const int r = lt >> 1;
    const int c = (lt & 1) * 16;
#if !HAVE_TDM
    h8 q0 = *(const h8*)(qg + hbase + r * 32 + c);
    h8 q1 = *(const h8*)(qg + hbase + r * 32 + c + 8);
    *(h8*)(&sQ[side][r * 40 + c])     = q0;
    *(h8*)(&sQ[side][r * 40 + c + 8]) = q1;
    h8 k0 = *(const h8*)(kg + hbase + r * 32 + c);
    h8 k1 = *(const h8*)(kg + hbase + r * 32 + c + 8);
    *(h8*)(&sK[side][r * 40 + c])     = k0;
    *(h8*)(&sK[side][r * 40 + c + 8]) = k1;
#endif
    h8 v0 = *(const h8*)(vg + hbase + r * 32 + c);
    h8 v1 = *(const h8*)(vg + hbase + r * 32 + c + 8);
#pragma unroll
    for (int t2 = 0; t2 < 8; ++t2) {
      sVT[side][(c + t2)     * 72 + r] = v0[t2];
      sVT[side][(c + 8 + t2) * 72 + r] = v1[t2];
    }
#if HAVE_TDM
    if (wv == 0) __builtin_amdgcn_s_wait_tensorcnt(0);
#endif
  }
  __syncthreads();

  // scores = q @ k^T (+ pos_bias + mask), 16x16 tiles, K = 32
  {
    const int m0  = wv * 16;
    const int ar  = m0 + (lane & 15);
    const int akh = (lane >> 4) << 3;
    h8 a0 = *(const h8*)(&sQ[side][ar * 40 + akh]);
    h8 a1 = *(const h8*)(&sQ[side][ar * 40 + akh + 16]);
    h16 av = cat16(a0, a1);
#pragma unroll
    for (int nt = 0; nt < 4; ++nt) {
      const int n0  = nt * 16;
      const int bc  = n0 + (lane & 15);
      const int bk0 = (lane >> 4) << 3;
      h8 b0 = *(const h8*)(&sK[side][bc * 40 + bk0]);
      h8 b1 = *(const h8*)(&sK[side][bc * 40 + bk0 + 16]);
      f8 c8 = {0.f,0.f,0.f,0.f,0.f,0.f,0.f,0.f};
      c8 = __builtin_amdgcn_wmma_f32_16x16x32_f16(false, av, false, cat16(b0, b1),
                                                  (short)0, c8, false, false);
      const int sm   = m0 + ((lane >> 4) * 8);
      const int tcol = n0 + (lane & 15);
#pragma unroll
      for (int i = 0; i < 8; ++i) {
        int srow = sm + i;
        float val = c8[i] + pb[((size_t)h * 64 + srow) * 64 + tcol]
                          + mask[((size_t)w * 64 + srow) * 64 + tcol];
        sSC[side][srow * 68 + tcol] = val;
      }
    }
  }
  __syncthreads();

  // softmax per row; probs written as f16 in-place behind the f32 read cursor
  if (tid < 128) {
    const int sd = tid >> 6, row = tid & 63;
    float* rp = &sSC[sd][row * 68];
    float mx = -1e30f;
    for (int t2 = 0; t2 < 64; ++t2) mx = fmaxf(mx, rp[t2]);
    float sum = 0.f;
    for (int t2 = 0; t2 < 64; ++t2) sum += __expf(rp[t2] - mx);
    float inv = 1.0f / sum;
    _Float16* hp2 = (_Float16*)rp;
    for (int t2 = 0; t2 < 64; ++t2) {
      float e2 = __expf(rp[t2] - mx) * inv;
      hp2[t2] = (_Float16)e2;
    }
  }
  __syncthreads();

  // out = probs @ v (K = 64 -> two WMMAs per 16x16 tile)
  {
    const int m0 = wv * 16;
    const int ar = m0 + (lane & 15);
    const _Float16* prow = (const _Float16*)&sSC[side][ar * 68];
#pragma unroll
    for (int nt = 0; nt < 2; ++nt) {
      f8 c8 = {0.f,0.f,0.f,0.f,0.f,0.f,0.f,0.f};
#pragma unroll
      for (int kc = 0; kc < 2; ++kc) {
        const int kbase = kc * 32;
        const int akh = kbase + ((lane >> 4) << 3);
        h8 a0 = *(const h8*)(prow + akh);
        h8 a1 = *(const h8*)(prow + akh + 16);
        const int bc  = nt * 16 + (lane & 15);
        const int bk0 = kbase + ((lane >> 4) << 3);
        h8 b0 = *(const h8*)(&sVT[side][bc * 72 + bk0]);
        h8 b1 = *(const h8*)(&sVT[side][bc * 72 + bk0 + 16]);
        c8 = __builtin_amdgcn_wmma_f32_16x16x32_f16(false, cat16(a0, a1),
                                                    false, cat16(b0, b1),
                                                    (short)0, c8, false, false);
      }
      const int sm = m0 + ((lane >> 4) * 8);
      const int dh = nt * 16 + (lane & 15);
#pragma unroll
      for (int i = 0; i < 8; ++i) {
        int srow = sm + i;
        z[(((size_t)bw * Sdim) + srow) * Edim + h * DH + dh] = (_Float16)c8[i];
      }
    }
  }
}

// ---------------------------------------------------------------------------
// K4: output projection + window reverse + roll(+4,+4) + residual add into xt
// ---------------------------------------------------------------------------
__global__ __launch_bounds__(256)
void k_proj(const _Float16* __restrict__ z, const float* __restrict__ Wo,
            const float* __restrict__ bo, float* __restrict__ xt)
{
  __shared__ __align__(16) _Float16 ldsA[64 * 40];
  __shared__ __align__(16) _Float16 ldsBT[64 * 40];

  const int m0 = blockIdx.x * 64;
  const int n0 = blockIdx.y * 64;
  AccPair r = gemm_core(z, Wo, Edim, Edim, m0, n0, ldsA, ldsBT);

  const int lane = threadIdx.x & 31;
  const int wid  = threadIdx.x >> 5;
  const int wm = wid & 3, wn2 = wid >> 2;
  const int mbase = m0 + wm * 16 + ((lane >> 4) * 8);
#pragma unroll
  for (int nt = 0; nt < 2; ++nt) {
    f8 acc = nt ? r.a1 : r.a0;
    const int gn = n0 + wn2 * 32 + nt * 16 + (lane & 15);
    const float bb = bo[gn];
#pragma unroll
    for (int i = 0; i < 8; ++i) {
      int gm = mbase + i;
      int bw = gm >> 6, sIdx = gm & 63;
      int b = bw >> 6, w = bw & 63;
      int ii = (w >> 3) * 8 + (sIdx >> 3);
      int jj = (w & 7) * 8 + (sIdx & 7);
      int d1 = (ii + 4) & 63, d2 = (jj + 4) & 63;
      size_t tok = (size_t)b * 4096 + (size_t)d1 * 64 + d2;
      xt[tok * Edim + gn] += acc[i] + bb;
    }
  }
}

// ---------------------------------------------------------------------------
// K5: LayerNorm2 -> f16.  One wave per token; shuffle reduction.
// ---------------------------------------------------------------------------
__global__ __launch_bounds__(256)
void k_ln2(const float* __restrict__ xt, const float* __restrict__ g2,
           const float* __restrict__ b2, _Float16* __restrict__ hh)
{
  const size_t tok = (size_t)blockIdx.x * 8 + (threadIdx.x >> 5);
  const int lane = threadIdx.x & 31;
  const float* row = xt + tok * Edim;
  float s = 0.f, sq = 0.f;
  float vals[12];
#pragma unroll
  for (int k2 = 0; k2 < 12; ++k2) {
    float v = row[lane + 32 * k2];
    vals[k2] = v; s += v; sq += v * v;
  }
#pragma unroll
  for (int off = 16; off > 0; off >>= 1) {
    s  += __shfl_xor(s,  off, 32);
    sq += __shfl_xor(sq, off, 32);
  }
  float mu  = s * (1.0f / Edim);
  float var = sq * (1.0f / Edim) - mu * mu;
  float rs  = rsqrtf(var + 1e-5f);
  _Float16* orow = hh + tok * Edim;
#pragma unroll
  for (int k2 = 0; k2 < 12; ++k2) {
    int e = lane + 32 * k2;
    orow[e] = (_Float16)((vals[k2] - mu) * rs * g2[e] + b2[e]);
  }
}

// ---------------------------------------------------------------------------
// K6: MLP fc1 + GELU -> f16 hidden
// ---------------------------------------------------------------------------
__global__ __launch_bounds__(256)
void k_mlp1(const _Float16* __restrict__ hh, const float* __restrict__ Wm1,
            const float* __restrict__ bm1, _Float16* __restrict__ y1)
{
  __shared__ __align__(16) _Float16 ldsA[64 * 40];
  __shared__ __align__(16) _Float16 ldsBT[64 * 40];

  const int m0 = blockIdx.x * 64;
  const int n0 = blockIdx.y * 64;
  AccPair r = gemm_core(hh, Wm1, Edim, HID, m0, n0, ldsA, ldsBT);

  const int lane = threadIdx.x & 31;
  const int wid  = threadIdx.x >> 5;
  const int wm = wid & 3, wn2 = wid >> 2;
  const int mbase = m0 + wm * 16 + ((lane >> 4) * 8);
#pragma unroll
  for (int nt = 0; nt < 2; ++nt) {
    f8 acc = nt ? r.a1 : r.a0;
    const int gn = n0 + wn2 * 32 + nt * 16 + (lane & 15);
    const float bb = bm1[gn];
#pragma unroll
    for (int i = 0; i < 8; ++i) {
      int gm = mbase + i;
      y1[(size_t)gm * HID + gn] = (_Float16)gelu_exact(acc[i] + bb);
    }
  }
}

// ---------------------------------------------------------------------------
// K7: MLP fc2 + GELU + residual, stored channels-first f32
// ---------------------------------------------------------------------------
__global__ __launch_bounds__(256)
void k_mlp2(const _Float16* __restrict__ y1, const float* __restrict__ Wm2,
            const float* __restrict__ bm2, const float* __restrict__ xt,
            float* __restrict__ out)
{
  __shared__ __align__(16) _Float16 ldsA[64 * 40];
  __shared__ __align__(16) _Float16 ldsBT[64 * 40];

  const int m0 = blockIdx.x * 64;
  const int n0 = blockIdx.y * 64;
  AccPair r = gemm_core(y1, Wm2, HID, Edim, m0, n0, ldsA, ldsBT);

  const int lane = threadIdx.x & 31;
  const int wid  = threadIdx.x >> 5;
  const int wm = wid & 3, wn2 = wid >> 2;
  const int mbase = m0 + wm * 16 + ((lane >> 4) * 8);
#pragma unroll
  for (int nt = 0; nt < 2; ++nt) {
    f8 acc = nt ? r.a1 : r.a0;
    const int gn = n0 + wn2 * 32 + nt * 16 + (lane & 15);
    const float bb = bm2[gn];
#pragma unroll
    for (int i = 0; i < 8; ++i) {
      int gm = mbase + i;
      float val = gelu_exact(acc[i] + bb) + xt[(size_t)gm * Edim + gn];
      int b = gm >> 12, pos = gm & 4095;
      out[((size_t)b * Edim + gn) * 4096 + pos] = val;
    }
  }
}

// ---------------------------------------------------------------------------
extern "C" void kernel_launch(void* const* d_in, const int* in_sizes, int n_in,
                              void* d_out, int out_size, void* d_ws, size_t ws_size,
                              hipStream_t stream)
{
  (void)in_sizes; (void)n_in; (void)out_size; (void)ws_size;
  const float* x    = (const float*)d_in[0];
  const float* mask = (const float*)d_in[1];
  const float* g1   = (const float*)d_in[2];
  const float* b1   = (const float*)d_in[3];
  const float* Wq   = (const float*)d_in[4];
  const float* bq   = (const float*)d_in[5];
  const float* Wk   = (const float*)d_in[6];
  const float* bk   = (const float*)d_in[7];
  const float* Wv   = (const float*)d_in[8];
  const float* bv   = (const float*)d_in[9];
  const float* Wo   = (const float*)d_in[10];
  const float* bo   = (const float*)d_in[11];
  const float* pb   = (const float*)d_in[12];
  const float* g2   = (const float*)d_in[13];
  const float* b2   = (const float*)d_in[14];
  const float* Wm1  = (const float*)d_in[15];
  const float* bm1  = (const float*)d_in[16];
  const float* Wm2  = (const float*)d_in[17];
  const float* bm2  = (const float*)d_in[18];
  float* out = (float*)d_out;

  char* w8 = (char*)d_ws;
  float*    xt = (float*)w8;                                   // NTOK*384 f32
  _Float16* xw = (_Float16*)(w8 + (size_t)NTOK * Edim * 4);    // NTOK*384 f16
  _Float16* qh = xw + (size_t)NTOK * Edim;
  _Float16* kh = qh + (size_t)NTOK * Edim;
  _Float16* vh = kh + (size_t)NTOK * Edim;
  _Float16* zh = vh + (size_t)NTOK * Edim;
  _Float16* hh = zh + (size_t)NTOK * Edim;
  _Float16* y1 = hh + (size_t)NTOK * Edim;                     // NTOK*1536 f16

  k_ln1_window<<<Bdim * Ddim, 256, 0, stream>>>(x, g1, b1, xt, xw);
  k_qkv <<<dim3(NTOK / 64, Edim / 64, 3), 256, 0, stream>>>(xw, Wq, bq, Wk, bk,
                                                            Wv, bv, qh, kh, vh);
  k_attn<<<Bdim * NWdim * (Hnum / 2), 256, 0, stream>>>(qh, kh, vh, pb, mask, zh);
  k_proj<<<dim3(NTOK / 64, Edim / 64), 256, 0, stream>>>(zh, Wo, bo, xt);
  k_ln2 <<<NTOK / 8, 256, 0, stream>>>(xt, g2, b2, hh);
  k_mlp1<<<dim3(NTOK / 64, HID / 64), 256, 0, stream>>>(hh, Wm1, bm1, y1);
  k_mlp2<<<dim3(NTOK / 64, Edim / 64), 256, 0, stream>>>(y1, Wm2, bm2, xt, out);
}
